// ErwinEmbedding_21002390077514
// MI455X (gfx1250) — compile-verified, matrix operand order checked by reading, not run on hardware
//
#include <hip/hip_runtime.h>
#include <hip/hip_bf16.h>

#define DIM   128
#define KMSG  288   // 2*DIM + 3 = 259, padded to 9 chunks of 32
#define KUPD  256
#define KEMB  64
#define LN_EPS 1e-5f

typedef __attribute__((ext_vector_type(16))) _Float16 v16h;
typedef __attribute__((ext_vector_type(8)))  _Float16 v8h;
typedef __attribute__((ext_vector_type(8)))  float    v8f;

// Load a 16x32 f16 WMMA A/B fragment for this lane.
// Per CDNA5 ISA 16-bit A layout: lane half 0 holds K = {0..7, 16..23},
// half 1 holds K = {8..15, 24..31} -> two contiguous 8xf16 (16B) loads.
// Caller passes p = rowBase + chunk*32 + half*8 (16B aligned).
__device__ __forceinline__ v16h frag16(const _Float16* p) {
  v8h lo = *(const v8h*)(p);
  v8h hi = *(const v8h*)(p + 16);
  return __builtin_shufflevector(lo, hi, 0,1,2,3,4,5,6,7,8,9,10,11,12,13,14,15);
}

__device__ __forceinline__ v8f wmma16(v16h a, v16h b, v8f c) {
  return __builtin_amdgcn_wmma_f32_16x16x32_f16(false, a, false, b, (short)0, c, false, false);
}

// Cooperatively stage `bytes` of weights from global into LDS using the CDNA5
// async global->LDS copy path (ASYNCcnt-tracked), then sync the workgroup.
// LDS address = low 32 bits of the generic pointer (LDS aperture keeps the
// offset in addr[31:0]). All 256 threads must be active (called pre-guard).
__device__ __forceinline__ void stage_to_lds(const _Float16* __restrict__ g,
                                             _Float16* s, int bytes) {
  const uint32_t base = (uint32_t)(uintptr_t)s;
  const char* gp = (const char*)g;
  for (int i = (int)threadIdx.x * 16; i < bytes; i += 256 * 16) {
    const uint32_t lds = base + (uint32_t)i;
    const char* ga = gp + i;
    asm volatile("global_load_async_to_lds_b128 %0, %1, off"
                 :: "v"(lds), "v"(ga) : "memory");
  }
  asm volatile("s_wait_asynccnt 0" ::: "memory");
  __syncthreads();
}

// ---------------- conversion / prep kernels ----------------

__global__ void k_cvt_f16(const float* __restrict__ src, _Float16* __restrict__ dst, int n) {
  int i = blockIdx.x * 256 + threadIdx.x;
  if (i < n) dst[i] = (_Float16)src[i];
}

// dst[(s*128 + n)*KP + k] = (k < K) ? src[(s*K + k)*128 + n] : 0   (B^T, K-major, padded)
__global__ void k_transpose_pad(const float* __restrict__ src, _Float16* __restrict__ dst,
                                int S, int K, int KP) {
  int i = blockIdx.x * 256 + threadIdx.x;
  int total = S * 128 * KP;
  if (i >= total) return;
  int k = i % KP;
  int n = (i / KP) % 128;
  int s = i / (KP * 128);
  dst[i] = (k < K) ? (_Float16)src[((size_t)s * K + k) * 128 + n] : (_Float16)0.0f;
}

__global__ void k_degree(const int* __restrict__ ei, float* __restrict__ counts, int E) {
  int i = blockIdx.x * 256 + threadIdx.x;
  if (i < E) atomicAdd(&counts[ei[E + i]], 1.0f);
}

// ---------------- embed: h = x @ W_e + b  (N x 64 x 128) ----------------

__global__ void k_embed(const _Float16* __restrict__ x16, const _Float16* __restrict__ wT,
                        const float* __restrict__ bias, float* __restrict__ h,
                        _Float16* __restrict__ h16, int nTiles) {
  const int lane = threadIdx.x & 31;
  const int wave = threadIdx.x >> 5;
  const int tile = blockIdx.x * 8 + wave;
  if (tile >= nTiles) return;                 // wave-uniform: EXEC stays all-ones for WMMA
  const int ml = lane & 15, half = lane >> 4;
  const int node = tile * 16 + ml;

  v8f acc[8];
#pragma unroll
  for (int n = 0; n < 8; ++n)
#pragma unroll
    for (int r = 0; r < 8; ++r) acc[n][r] = 0.0f;

  const _Float16* arow = x16 + (size_t)node * KEMB;
#pragma unroll
  for (int c = 0; c < 2; ++c) {
    v16h a = frag16(arow + c * 32 + half * 8);
#pragma unroll
    for (int n = 0; n < 8; ++n) {
      const _Float16* bp = wT + (size_t)(n * 16 + ml) * KEMB + c * 32 + half * 8;
      acc[n] = wmma16(a, frag16(bp), acc[n]);
    }
  }
#pragma unroll
  for (int n = 0; n < 8; ++n) {
    const int colI = n * 16 + ml;
    const float bv = bias[colI];
#pragma unroll
    for (int r = 0; r < 8; ++r) {
      const int node2 = tile * 16 + r + half * 8;   // C layout: M = r + half*8, N = n*16+ml
      const float v = acc[n][r] + bv;
      h  [(size_t)node2 * DIM + colI] = v;
      h16[(size_t)node2 * DIM + colI] = (_Float16)v;
    }
  }
}

// ---------------- message: fused gather + GEMM + GELU + LN + atomic scatter ----------------

__global__ void k_message(const _Float16* __restrict__ h16, const float* __restrict__ pos,
                          const int* __restrict__ ei, const _Float16* __restrict__ wT,
                          const float* __restrict__ bias, float* __restrict__ agg, int E) {
  // Stage the 288x128 f16 weight tile (72 KB, > 64 KB WGP$) into LDS once per
  // block via async copies; 8 waves then stream B-fragments from LDS.
  __shared__ __attribute__((aligned(16))) _Float16 sB[128 * KMSG];
  stage_to_lds(wT, sB, 128 * KMSG * 2);

  const int lane = threadIdx.x & 31;
  const int wave = threadIdx.x >> 5;
  const int tile = blockIdx.x * 8 + wave;
  const int nT = E >> 4;
  if (tile >= nT) return;
  const int ml = lane & 15, half = lane >> 4;
  const int e = tile * 16 + ml;
  const int row = ei[e];
  const int col = ei[E + e];

  v8f acc[8];
#pragma unroll
  for (int n = 0; n < 8; ++n)
#pragma unroll
    for (int r = 0; r < 8; ++r) acc[n][r] = 0.0f;

  const _Float16* hr = h16 + (size_t)row * DIM;
  const _Float16* hc = h16 + (size_t)col * DIM;

#pragma unroll
  for (int c = 0; c < 9; ++c) {
    v16h a;
    if (c < 4) {
      a = frag16(hr + c * 32 + half * 8);
    } else if (c < 8) {
      a = frag16(hc + (c - 4) * 32 + half * 8);
    } else {
      // K chunk 8: edge_attr in K-local 0..2 (held by half==0 lanes, elems 0..2), rest zero
#pragma unroll
      for (int j = 0; j < 16; ++j) a[j] = (_Float16)0.0f;
      if (half == 0) {
        a[0] = (_Float16)(pos[row * 3 + 0] - pos[col * 3 + 0]);
        a[1] = (_Float16)(pos[row * 3 + 1] - pos[col * 3 + 1]);
        a[2] = (_Float16)(pos[row * 3 + 2] - pos[col * 3 + 2]);
      }
    }
#pragma unroll
    for (int n = 0; n < 8; ++n) {
      const _Float16* bp = sB + (size_t)(n * 16 + ml) * KMSG + c * 32 + half * 8;
      acc[n] = wmma16(a, frag16(bp), acc[n]);
    }
  }

  // bias + exact GELU
#pragma unroll
  for (int n = 0; n < 8; ++n) {
    const float bv = bias[n * 16 + ml];
#pragma unroll
    for (int r = 0; r < 8; ++r) {
      float v = acc[n][r] + bv;
      v = 0.5f * v * (1.0f + erff(v * 0.70710678118654752f));
      acc[n][r] = v;
    }
  }

  // per-edge LayerNorm (row m = r + half*8 lives in one 16-lane half) + scatter-add
#pragma unroll
  for (int r = 0; r < 8; ++r) {
    float s = 0.f, q = 0.f;
#pragma unroll
    for (int n = 0; n < 8; ++n) { const float v = acc[n][r]; s += v; q += v * v; }
#pragma unroll
    for (int off = 1; off < 16; off <<= 1) {
      s += __shfl_xor(s, off, 32);
      q += __shfl_xor(q, off, 32);
    }
    const float mean = s * (1.0f / 128.0f);
    const float var  = q * (1.0f / 128.0f) - mean * mean;
    const float rstd = rsqrtf(var + LN_EPS);
    const int   cm   = __shfl(col, r + half * 8, 32);   // col index of row m's edge
    float* dst = agg + (size_t)cm * DIM;
#pragma unroll
    for (int n = 0; n < 8; ++n) {
      const float v = (acc[n][r] - mean) * rstd;
      atomicAdd(dst + n * 16 + ml, v);
    }
  }
}

// ---------------- update: h += LN([h, agg/denom] @ W_u + b) ----------------

__global__ void k_update(float* __restrict__ h, _Float16* __restrict__ h16,
                         const float* __restrict__ agg, const float* __restrict__ counts,
                         const _Float16* __restrict__ wT, const float* __restrict__ bias,
                         int nTiles) {
  __shared__ __attribute__((aligned(16))) _Float16 sB[128 * KUPD];   // 64 KB
  stage_to_lds(wT, sB, 128 * KUPD * 2);

  const int lane = threadIdx.x & 31;
  const int wave = threadIdx.x >> 5;
  const int tile = blockIdx.x * 8 + wave;
  if (tile >= nTiles) return;
  const int ml = lane & 15, half = lane >> 4;
  const int node = tile * 16 + ml;
  const float invd = 1.0f / fmaxf(counts[node], 1.0f);

  v8f acc[8];
#pragma unroll
  for (int n = 0; n < 8; ++n)
#pragma unroll
    for (int r = 0; r < 8; ++r) acc[n][r] = 0.0f;

  const _Float16* hrow = h16 + (size_t)node * DIM;
  const float*    arow = agg + (size_t)node * DIM;

#pragma unroll
  for (int c = 0; c < 8; ++c) {
    v16h a;
    if (c < 4) {
      a = frag16(hrow + c * 32 + half * 8);
    } else {
      const float* p = arow + (c - 4) * 32 + half * 8;
#pragma unroll
      for (int j = 0; j < 8; ++j) {
        a[j]     = (_Float16)(p[j]      * invd);
        a[8 + j] = (_Float16)(p[16 + j] * invd);
      }
    }
#pragma unroll
    for (int n = 0; n < 8; ++n) {
      const _Float16* bp = sB + (size_t)(n * 16 + ml) * KUPD + c * 32 + half * 8;
      acc[n] = wmma16(a, frag16(bp), acc[n]);
    }
  }

#pragma unroll
  for (int n = 0; n < 8; ++n) {
    const float bv = bias[n * 16 + ml];
#pragma unroll
    for (int r = 0; r < 8; ++r) acc[n][r] += bv;
  }

#pragma unroll
  for (int r = 0; r < 8; ++r) {
    float s = 0.f, q = 0.f;
#pragma unroll
    for (int n = 0; n < 8; ++n) { const float v = acc[n][r]; s += v; q += v * v; }
#pragma unroll
    for (int off = 1; off < 16; off <<= 1) {
      s += __shfl_xor(s, off, 32);
      q += __shfl_xor(q, off, 32);
    }
    const float mean = s * (1.0f / 128.0f);
    const float var  = q * (1.0f / 128.0f) - mean * mean;
    const float rstd = rsqrtf(var + LN_EPS);
    const int node2 = tile * 16 + r + half * 8;
#pragma unroll
    for (int n = 0; n < 8; ++n) {
      const int colI = n * 16 + ml;
      const float v  = (acc[n][r] - mean) * rstd;
      const float hv = h[(size_t)node2 * DIM + colI] + v;
      h  [(size_t)node2 * DIM + colI] = hv;
      h16[(size_t)node2 * DIM + colI] = (_Float16)hv;
    }
  }
}

// ---------------- host launcher ----------------

extern "C" void kernel_launch(void* const* d_in, const int* in_sizes, int n_in,
                              void* d_out, int out_size, void* d_ws, size_t ws_size,
                              hipStream_t stream) {
  const float* x       = (const float*)d_in[0];
  const float* pos     = (const float*)d_in[1];
  const int*   ei      = (const int*)  d_in[2];
  const float* embed_w = (const float*)d_in[3];
  const float* embed_b = (const float*)d_in[4];
  const float* msg_w   = (const float*)d_in[5];
  const float* msg_b   = (const float*)d_in[6];
  const float* upd_w   = (const float*)d_in[7];
  const float* upd_b   = (const float*)d_in[8];
  float* h = (float*)d_out;

  const int N = in_sizes[0] / 64;   // 50000
  const int E = in_sizes[2] / 2;    // 800000

  char* p = (char*)d_ws;
  auto carve = [&](size_t bytes) { char* r = p; p += (bytes + 255) & ~(size_t)255; return r; };
  _Float16* h16  = (_Float16*)carve((size_t)N * DIM * 2);
  float*    agg  = (float*)   carve((size_t)N * DIM * 4);
  float*    cnts = (float*)   carve((size_t)N * 4);
  _Float16* x16  = (_Float16*)carve((size_t)N * KEMB * 2);
  _Float16* ewT  = (_Float16*)carve((size_t)128 * KEMB * 2);
  _Float16* mwT  = (_Float16*)carve((size_t)3 * 128 * KMSG * 2);
  _Float16* uwT  = (_Float16*)carve((size_t)3 * 128 * KUPD * 2);

  { int n = N * KEMB;        k_cvt_f16      <<<(n + 255) / 256, 256, 0, stream>>>(x, x16, n); }
  { int t = 1 * 128 * KEMB;  k_transpose_pad<<<(t + 255) / 256, 256, 0, stream>>>(embed_w, ewT, 1,  64, KEMB); }
  { int t = 3 * 128 * KMSG;  k_transpose_pad<<<(t + 255) / 256, 256, 0, stream>>>(msg_w,   mwT, 3, 259, KMSG); }
  { int t = 3 * 128 * KUPD;  k_transpose_pad<<<(t + 255) / 256, 256, 0, stream>>>(upd_w,   uwT, 3, 256, KUPD); }

  hipMemsetAsync(cnts, 0, (size_t)N * 4, stream);
  k_degree<<<(E + 255) / 256, 256, 0, stream>>>(ei, cnts, E);

  const int nTiles = N / 16;   // 3125 exact
  const int eTiles = E / 16;   // 50000 exact
  k_embed<<<(nTiles + 7) / 8, 256, 0, stream>>>(x16, ewT, embed_b, h, h16, nTiles);

  for (int s = 0; s < 3; ++s) {
    hipMemsetAsync(agg, 0, (size_t)N * DIM * 4, stream);
    k_message<<<(eTiles + 7) / 8, 256, 0, stream>>>(h16, pos, ei,
                                                    mwT + (size_t)s * 128 * KMSG,
                                                    msg_b + (size_t)s * DIM, agg, E);
    k_update<<<(nTiles + 7) / 8, 256, 0, stream>>>(h, h16, agg, cnts,
                                                   uwT + (size_t)s * 128 * KUPD,
                                                   upd_b + (size_t)s * DIM, nTiles);
  }
}
